// DbrxAttention_40492951667583
// MI455X (gfx1250) — matrix operand order repaired and checked by reading, not compiled, not run on hardware
//
#include <hip/hip_runtime.h>

typedef __bf16 bf16_t;
typedef __attribute__((ext_vector_type(8)))  __bf16 v8bf;
typedef __attribute__((ext_vector_type(16))) __bf16 v16bf;
typedef __attribute__((ext_vector_type(8)))  float  v8f;
typedef __attribute__((ext_vector_type(4)))  float  v4f;

#define S_LEN   2048
#define D_MODEL 6144
#define NH      48
#define NKV     8
#define HD      128
#define NQKV    8192   // (48 + 2*8) * 128

static __device__ __forceinline__ v16bf cat16(v8bf lo, v8bf hi) {
  return __builtin_shufflevector(lo, hi, 0,1,2,3,4,5,6,7,8,9,10,11,12,13,14,15);
}

// ---- CDNA5 async global->LDS copy (ASYNCcnt path), with safe fallback -----
typedef __attribute__((__vector_size__(16))) int v4i_vs;
typedef __attribute__((address_space(1))) v4i_vs gv4i;   // global b128 payload
typedef __attribute__((address_space(3))) v4i_vs lv4i;   // LDS b128 payload

#if __has_builtin(__builtin_amdgcn_global_load_async_to_lds_b128) && \
    __has_builtin(__builtin_amdgcn_s_wait_asynccnt)
#define HAVE_ASYNC_LDS 1
#endif

static __device__ __forceinline__ void async_copy16(void* lds, const void* g) {
#ifdef HAVE_ASYNC_LDS
  __builtin_amdgcn_global_load_async_to_lds_b128(
      (gv4i*)(uintptr_t)g, (lv4i*)(unsigned)(uintptr_t)lds, 0, 0);
#else
  *(v8bf*)lds = *(const v8bf*)g;
#endif
}
static __device__ __forceinline__ void async_wait() {
#ifdef HAVE_ASYNC_LDS
  __builtin_amdgcn_s_wait_asynccnt(0);
#endif
}

// ---------------------------------------------------------------------------
// f32 -> bf16 cast (8 elements per thread)
// ---------------------------------------------------------------------------
__global__ __launch_bounds__(256)
void cast_f32_bf16(const float* __restrict__ src, bf16_t* __restrict__ dst) {
  const size_t i = (size_t)(blockIdx.x * 256 + threadIdx.x) * 8;
  v4f f0 = *(const v4f*)(src + i);
  v4f f1 = *(const v4f*)(src + i + 4);
  v8bf h;
  #pragma unroll
  for (int e = 0; e < 4; ++e) { h[e] = (__bf16)f0[e]; h[4 + e] = (__bf16)f1[e]; }
  *(v8bf*)(dst + i) = h;
}

// ---------------------------------------------------------------------------
// GEMM: C[MxN] = A[MxK] * B[KxN], bf16 in, f32 out, bf16 WMMA math.
// Block: 256 threads (8 waves, 4Mx2N). C tile 128x128, K-step 32.
// A tile staged via async global->LDS; B tile transposed by hand.
// ---------------------------------------------------------------------------
__global__ __launch_bounds__(256)
void gemm_bf16wmma(const bf16_t* __restrict__ A, const bf16_t* __restrict__ B,
                   float* __restrict__ C, int M, int N, int K) {
  __shared__ alignas(16) bf16_t sA[128][40];    // 128 m x 32 k (padded)
  __shared__ alignas(16) bf16_t sBt[128][40];   // 128 n x 32 k (transposed)

  const int t = threadIdx.x;
  const int w = t >> 5;
  const int L = t & 31;
  const int lane16 = L & 15;
  const int hiL = (L >> 4) & 1;
  const int wm = w >> 1;            // 0..3  (M direction, 32 rows each)
  const int wn = w & 1;             // 0..1  (N direction, 64 cols each)
  const int m0 = blockIdx.y * 128;
  const int n0 = blockIdx.x * 128;
  const int base8  = hiL * 8;
  const int base16 = hiL * 16;

  const v8f vzero = {0.f,0.f,0.f,0.f,0.f,0.f,0.f,0.f};
  v8f acc[2][4];
  #pragma unroll
  for (int i = 0; i < 2; ++i)
    #pragma unroll
    for (int j = 0; j < 4; ++j) acc[i][j] = vzero;

  const int bn = t & 127;
  const int bkh = (t >> 7) * 16;

  for (int k0 = 0; k0 < K; k0 += 32) {
    // Stage A tile (128x32 bf16) via async copy: 512 x 16B chunks
    #pragma unroll
    for (int i = 0; i < 2; ++i) {
      const int ch = t + i * 256;
      const int r = ch >> 2, c = ch & 3;
      async_copy16(&sA[r][c * 8], A + (size_t)(m0 + r) * K + k0 + c * 8);
    }
    // Stage B tile (32x128) transposed -> sBt[n][k]
    {
      const bf16_t* bp = B + (size_t)(k0 + bkh) * N + n0 + bn;
      if (k0 + 32 < K) __builtin_prefetch(bp + (size_t)32 * N, 0, 1);
      v8bf h0, h1;
      #pragma unroll
      for (int e = 0; e < 8; ++e) {
        h0[e] = bp[(size_t)e * N];
        h1[e] = bp[(size_t)(8 + e) * N];
      }
      *(v8bf*)&sBt[bn][bkh]     = h0;
      *(v8bf*)&sBt[bn][bkh + 8] = h1;
    }
    async_wait();
    __syncthreads();

    // A fragments (rows wm*32 .. +31), reused across all 4 N sub-tiles
    v16bf af[2];
    #pragma unroll
    for (int i = 0; i < 2; ++i) {
      const int am = wm * 32 + i * 16 + lane16;
      af[i] = cat16(*(const v8bf*)&sA[am][base8],
                    *(const v8bf*)&sA[am][base8 + 16]);
    }
    #pragma unroll
    for (int j = 0; j < 4; ++j) {
      const int bnn = wn * 64 + j * 16 + lane16;
      v16bf bfrag = cat16(*(const v8bf*)&sBt[bnn][base8],
                          *(const v8bf*)&sBt[bnn][base8 + 16]);
      acc[0][j] = __builtin_amdgcn_wmma_f32_16x16x32_bf16(
          false, af[0], false, bfrag, (short)0, acc[0][j], false, false);
      acc[1][j] = __builtin_amdgcn_wmma_f32_16x16x32_bf16(
          false, af[1], false, bfrag, (short)0, acc[1][j], false, false);
    }
    __syncthreads();
  }

  #pragma unroll
  for (int i = 0; i < 2; ++i)
    #pragma unroll
    for (int j = 0; j < 4; ++j)
      #pragma unroll
      for (int r = 0; r < 8; ++r) {
        const int row = m0 + wm * 32 + i * 16 + r + hiL * 8;
        const int col = n0 + wn * 64 + j * 16 + lane16;
        C[(size_t)row * N + col] = acc[i][j][r];
      }
}

// ---------------------------------------------------------------------------
// RoPE + split QKV into head-major bf16 tensors.
// Qb: [H][S][HD], Kb/Vb: [KV][S][HD]
// ---------------------------------------------------------------------------
__global__ __launch_bounds__(256)
void rope_split(const float* __restrict__ qkv, const float* __restrict__ cosv,
                const float* __restrict__ sinv, bf16_t* __restrict__ Qb,
                bf16_t* __restrict__ Kb, bf16_t* __restrict__ Vb) {
  const int idx = blockIdx.x * 256 + threadIdx.x;
  const int s = idx >> 13;
  const int c = idx & 8191;
  const float* row = qkv + (size_t)s * NQKV;
  if (c < NH * HD) {
    const int h = c >> 7, d = c & 127;
    const float x  = row[c];
    const float xr = (d < 64) ? -row[c + 64] : row[c - 64];
    const float v  = x * cosv[s * HD + d] + xr * sinv[s * HD + d];
    Qb[((size_t)h * S_LEN + s) * HD + d] = (__bf16)v;
  } else if (c < (NH + NKV) * HD) {
    const int cc = c - NH * HD;
    const int kv = cc >> 7, d = cc & 127;
    const float x  = row[c];
    const float xr = (d < 64) ? -row[c + 64] : row[c - 64];
    const float v  = x * cosv[s * HD + d] + xr * sinv[s * HD + d];
    Kb[((size_t)kv * S_LEN + s) * HD + d] = (__bf16)v;
  } else {
    const int cc = c - (NH + NKV) * HD;
    const int kv = cc >> 7, d = cc & 127;
    Vb[((size_t)kv * S_LEN + s) * HD + d] = (__bf16)row[c];
  }
}

// ---------------------------------------------------------------------------
// Flash-attention, causal, GQA. Grid: (S/64, NH). Block: 128 = 4 waves.
// K staged via async global->LDS (shared by all 4 waves); V staged transposed.
// Output written as bf16 to feed the out-projection GEMM.
// ---------------------------------------------------------------------------
__global__ __launch_bounds__(128)
void attn_fa(const bf16_t* __restrict__ Qb, const bf16_t* __restrict__ Kb,
             const bf16_t* __restrict__ Vb, bf16_t* __restrict__ O) {
  __shared__ alignas(16) bf16_t sK[32][136];      // K tile: [key][hd] (padded)
  __shared__ alignas(16) bf16_t Vt[128][40];      // V tile transposed: [hd][key]
  __shared__ alignas(16) bf16_t sP[4][16][40];    // per-wave P (16q x 32k)

  const int t = threadIdx.x, w = t >> 5, L = t & 31;
  const int lane16 = L & 15, hiL = (L >> 4) & 1;
  const int h = blockIdx.y;
  const int kvh = h / (NH / NKV);
  const int q0 = blockIdx.x * 64 + w * 16;
  const float scale = 0.08838834764831845f;   // 1/sqrt(128)

  const bf16_t* Qh = Qb + (size_t)h   * S_LEN * HD;
  const bf16_t* Kh = Kb + (size_t)kvh * S_LEN * HD;
  const bf16_t* Vh = Vb + (size_t)kvh * S_LEN * HD;

  const int base8  = hiL * 8;
  const int base16 = hiL * 16;

  // Q fragments (A layout), register-resident
  v16bf qf[4];
  {
    const bf16_t* qrow = Qh + (size_t)(q0 + lane16) * HD;
    #pragma unroll
    for (int c = 0; c < 4; ++c)
      qf[c] = cat16(*(const v8bf*)(qrow + c * 32 + base8),
                    *(const v8bf*)(qrow + c * 32 + base8 + 16));
  }

  const v8f vzero = {0.f,0.f,0.f,0.f,0.f,0.f,0.f,0.f};
  v8f o[8];
  #pragma unroll
  for (int j = 0; j < 8; ++j) o[j] = vzero;
  float m[8], l[8];
  #pragma unroll
  for (int r = 0; r < 8; ++r) { m[r] = -1e30f; l[r] = 0.f; }

  const int nkb = blockIdx.x * 2 + 2;
  for (int kbi = 0; kbi < nkb; ++kbi) {
    const int kb = kbi * 32;

    // Stage K[kb..kb+32) via async copy: 512 x 16B chunks, 4 per thread
    #pragma unroll
    for (int i = 0; i < 4; ++i) {
      const int ch = t + i * 128;
      const int kr = ch >> 4, c = ch & 15;
      async_copy16(&sK[kr][c * 8], Kh + (size_t)(kb + kr) * HD + c * 8);
    }
    // Stage V[kb..kb+32) transposed
    {
      const int k   = t >> 2;
      const int hd0 = (t & 3) * 32;
      const bf16_t* vr = Vh + (size_t)(kb + k) * HD + hd0;
      #pragma unroll
      for (int i = 0; i < 32; ++i) Vt[hd0 + i][k] = vr[i];
    }
    async_wait();
    __syncthreads();

    // S = Q * K^T for two 16-key tiles (K fragments from LDS)
    v8f s0 = vzero, s1 = vzero;
    #pragma unroll
    for (int c = 0; c < 4; ++c) {
      v16bf k0f = cat16(*(const v8bf*)&sK[lane16][c * 32 + base8],
                        *(const v8bf*)&sK[lane16][c * 32 + base8 + 16]);
      s0 = __builtin_amdgcn_wmma_f32_16x16x32_bf16(
          false, qf[c], false, k0f, (short)0, s0, false, false);
      v16bf k1f = cat16(*(const v8bf*)&sK[16 + lane16][c * 32 + base8],
                        *(const v8bf*)&sK[16 + lane16][c * 32 + base8 + 16]);
      s1 = __builtin_amdgcn_wmma_f32_16x16x32_bf16(
          false, qf[c], false, k1f, (short)0, s1, false, false);
    }

    // Causal mask + online softmax
    const int key0 = kb + lane16, key1 = kb + 16 + lane16;
    #pragma unroll
    for (int r = 0; r < 8; ++r) {
      const int row = q0 + r + hiL * 8;
      float a0 = (key0 <= row) ? s0[r] * scale : -1e30f;
      float a1 = (key1 <= row) ? s1[r] * scale : -1e30f;
      float lm = fmaxf(a0, a1);
      #pragma unroll
      for (int d = 8; d >= 1; d >>= 1) lm = fmaxf(lm, __shfl_xor(lm, d, 16));
      const float nm = fmaxf(m[r], lm);
      const float alpha = __expf(m[r] - nm);
      const float p0 = __expf(a0 - nm);
      const float p1 = __expf(a1 - nm);
      l[r] = l[r] * alpha + p0 + p1;
      m[r] = nm;
      #pragma unroll
      for (int j = 0; j < 8; ++j) o[j][r] *= alpha;
      const int prow = r + hiL * 8;
      sP[w][prow][lane16]      = (__bf16)p0;
      sP[w][prow][16 + lane16] = (__bf16)p1;
    }

    // O += P * V
    v16bf pf = cat16(*(const v8bf*)&sP[w][lane16][base8],
                     *(const v8bf*)&sP[w][lane16][base8 + 16]);
    #pragma unroll
    for (int j = 0; j < 8; ++j) {
      v16bf vf = cat16(*(const v8bf*)&Vt[j * 16 + lane16][base8],
                       *(const v8bf*)&Vt[j * 16 + lane16][base8 + 16]);
      o[j] = __builtin_amdgcn_wmma_f32_16x16x32_bf16(
          false, pf, false, vf, (short)0, o[j], false, false);
    }
    __syncthreads();
  }

  // Normalize, convert to bf16, write O[row][h*128 + hd]
  #pragma unroll
  for (int r = 0; r < 8; ++r) {
    float ls = l[r];
    #pragma unroll
    for (int d = 8; d >= 1; d >>= 1) ls += __shfl_xor(ls, d, 16);
    const float inv = 1.0f / ls;
    const int row = q0 + r + hiL * 8;
    #pragma unroll
    for (int j = 0; j < 8; ++j)
      O[(size_t)row * D_MODEL + h * HD + j * 16 + lane16] =
          (__bf16)(o[j][r] * inv);
  }
}

// ---------------------------------------------------------------------------
extern "C" void kernel_launch(void* const* d_in, const int* in_sizes, int n_in,
                              void* d_out, int out_size, void* d_ws, size_t ws_size,
                              hipStream_t stream) {
  const float* hidden = (const float*)d_in[0];   // [1,2048,6144]
  const float* cosv   = (const float*)d_in[1];   // [2048,128]
  const float* sinv   = (const float*)d_in[2];   // [2048,128]
  const float* wqkv   = (const float*)d_in[3];   // [6144,8192]
  const float* wout   = (const float*)d_in[4];   // [6144,6144]
  float* out = (float*)d_out;                    // [1,2048,6144]

  char* ws = (char*)d_ws;
  bf16_t* Ah    = (bf16_t*)(ws);                           // 2048*6144 bf16
  bf16_t* Wq    = (bf16_t*)(ws + (size_t)25165824);        // 6144*8192 bf16
  bf16_t* Wo    = (bf16_t*)(ws + (size_t)125829120);       // 6144*6144 bf16
  float*  qkv   = (float*) (ws + (size_t)201326592);       // 2048*8192 f32
  bf16_t* Qb    = (bf16_t*)(ws + (size_t)268435456);       // 48*2048*128 bf16
  bf16_t* Kb    = (bf16_t*)(ws + (size_t)293601280);       //  8*2048*128 bf16
  bf16_t* Vb    = (bf16_t*)(ws + (size_t)297795584);       //  8*2048*128 bf16
  bf16_t* attnO = (bf16_t*)(ws + (size_t)301989888);       // 2048*6144 bf16

  // 0) cast activations + weights to bf16 (makes weight panels L2-resident)
  cast_f32_bf16<<<(S_LEN * D_MODEL) / 2048, 256, 0, stream>>>(hidden, Ah);
  cast_f32_bf16<<<(D_MODEL * NQKV) / 2048, 256, 0, stream>>>(wqkv, Wq);
  cast_f32_bf16<<<(D_MODEL * D_MODEL) / 2048, 256, 0, stream>>>(wout, Wo);
  // 1) QKV projection
  gemm_bf16wmma<<<dim3(NQKV / 128, S_LEN / 128), 256, 0, stream>>>(
      Ah, Wq, qkv, S_LEN, NQKV, D_MODEL);
  // 2) RoPE + split into head-major bf16
  rope_split<<<(S_LEN * NQKV) / 256, 256, 0, stream>>>(
      qkv, cosv, sinv, Qb, Kb, Vb);
  // 3) Causal flash attention
  attn_fa<<<dim3(S_LEN / 64, NH), 128, 0, stream>>>(Qb, Kb, Vb, attnO);
  // 4) Output projection
  gemm_bf16wmma<<<dim3(D_MODEL / 128, S_LEN / 128), 256, 0, stream>>>(
      attnO, Wo, out, S_LEN, D_MODEL, D_MODEL);
}
